// Transformer_6966436954790
// MI455X (gfx1250) — matrix operand order repaired and checked by reading, not compiled, not run on hardware
//
#include <hip/hip_runtime.h>
#include <hip/hip_bf16.h>
#include <cstdint>

// ---------------- problem constants ----------------
#define L_LAYERS 6
#define DMODEL   1024
#define NHEADS   16
#define KDIM     64
#define VOCAB    32000
#define BATCH    2
#define SEQ      2048
#define MTOT     (BATCH * SEQ)   // 4096 token rows
#define LN_EPS   1e-5f

typedef unsigned short ushort_t;

// ---------------- WMMA types ----------------
typedef __attribute__((ext_vector_type(16))) __bf16 v16bf;
typedef __attribute__((ext_vector_type(8)))  float  v8f;

union FragB {
    v16bf v;
    uint4 q[2];
};

// f32 -> bf16 (round to nearest even), bit-level
__device__ __forceinline__ ushort_t f2bf(float f) {
    unsigned u = __float_as_uint(f);
    u += 0x7FFFu + ((u >> 16) & 1u);
    return (ushort_t)(u >> 16);
}

// gfx1250 async DMA: copy 16B global -> LDS, tracked by ASYNCcnt
__device__ __forceinline__ void async_b128(const void* g, void* lds) {
    asm volatile("global_load_async_to_lds_b128 %0, %1, off"
                 :: "v"((unsigned)(uintptr_t)lds), "v"(g)
                 : "memory");
}
__device__ __forceinline__ void wait_async0() {
    asm volatile("s_wait_asynccnt 0x0" ::: "memory");
}

// =====================================================================
// Tiled transpose + f32->bf16 convert:  out[n*K + k] = bf16(in(k, n))
//   HEADED: in is [H][K][64] (QKV weights), n = h*64+kd -> in[h][k][kd]
//   else:   in is [K][N] row-major
//   grid: (N/32, K/32), 256 threads
// =====================================================================
template<bool HEADED>
__global__ __launch_bounds__(256)
void tconv_kernel(const float* __restrict__ in, ushort_t* __restrict__ out,
                  int K, int N)
{
    __shared__ ushort_t tile[32][33];
    const int n0 = blockIdx.x * 32, k0 = blockIdx.y * 32;
    const int c = threadIdx.x & 31, rr = threadIdx.x >> 5;   // 8 k-rows per pass
#pragma unroll
    for (int p = 0; p < 32; p += 8) {
        const int k = k0 + rr + p;
        const int n = n0 + c;
        float v;
        if (HEADED) v = in[(size_t)(n >> 6) * K * KDIM + (size_t)k * KDIM + (n & 63)];
        else        v = in[(size_t)k * N + n];
        tile[c][rr + p] = f2bf(v);
    }
    __syncthreads();
#pragma unroll
    for (int p = 0; p < 32; p += 8)
        out[(size_t)(n0 + rr + p) * K + k0 + c] = tile[rr + p][c];
}

// =====================================================================
// GEMM:  out = epilogue( A[M,K](bf16) * Bt[N,K](bf16, pre-transposed)^T )
//   Double-buffered LDS tiles filled by global_load_async_to_lds_b128.
//   BN==128: 8 waves as 2(M)x4(N), wave tile 64x32 (MT=4,NT=2)
//   BN==64 : 8 waves as 4(M)x2(N), wave tile 32x32 (MT=2,NT=2)
// Store modes:
//   SM_F32 : f32 out[row*ldc+col]
//   SM_BF16: bf16 out[row*ldc+col]
//   SM_QK  : bf16 out[((b*H+h)*S+s)*64 + kd] * storeScale
//   SM_V   : bf16 out[((b*H+h)*64+kd)*S + s] * storeScale (transposed)
// =====================================================================
#define SM_F32  0
#define SM_BF16 1
#define SM_QK   2
#define SM_V    3

template<int BN, int SMODE, bool HAS_BIAS, bool HAS_RES, bool RELU>
__global__ __launch_bounds__(256)
void gemm_bf16_wmma(const ushort_t* __restrict__ A, const ushort_t* __restrict__ Bt,
                    const float* __restrict__ bias, const float* __restrict__ res,
                    void* __restrict__ outv, int K, int ldc, float storeScale)
{
    constexpr int BM = 128, BK = 32;
    constexpr int KP = 40;                       // padded LDS K-stride: 80B, 16B-aligned
    constexpr int MT = (BN == 128) ? 4 : 2;
    constexpr int NT = 2;
    constexpr int WCOLS = (BN == 128) ? 4 : 2;

    __shared__ __align__(16) ushort_t As[2][BM * KP];
    __shared__ __align__(16) ushort_t Bs[2][BN * KP];

    const int tid  = threadIdx.x;
    const int lane = tid & 31;
    const int wid  = tid >> 5;
    const int waveN = wid % WCOLS;
    const int waveM = wid / WCOLS;

    const int blockRow = blockIdx.y * BM;
    const int colBase  = blockIdx.x * BN;

    const int mrow0 = waveM * (MT * 16);
    const int ncol0 = waveN * (NT * 16);
    const int r16 = lane & 15;
    const int kbA = (lane >> 4) * 8;    // A frag: K 0-7/16-23 vs 8-15/24-31
    const int kbB = (lane >> 4) * 8;    // B frag: K 0-7/16-23 vs 8-15/24-31

    // async staging maps (16B chunks)
    const int arow = tid >> 1, acb = (tid & 1) * 16;   // A: 2 chunks/thread
    const int brow = (BN == 128) ? (tid >> 1) : (tid >> 2);
    const int bcb  = (BN == 128) ? ((tid & 1) * 16) : ((tid & 3) * 8);

    const ushort_t* gA = A  + (size_t)(blockRow + arow) * K + acb;
    const ushort_t* gB = Bt + (size_t)(colBase + brow) * K + bcb;

    v8f acc[MT][NT];
#pragma unroll
    for (int i = 0; i < MT; ++i)
#pragma unroll
        for (int j = 0; j < NT; ++j)
#pragma unroll
            for (int r = 0; r < 8; ++r) acc[i][j][r] = 0.0f;

    const int KT = K / BK;

    // prologue: stage tile 0 into buffer 0
    async_b128(gA,     &As[0][arow * KP + acb]);
    async_b128(gA + 8, &As[0][arow * KP + acb + 8]);
    async_b128(gB,     &Bs[0][brow * KP + bcb]);
    if (BN == 128) async_b128(gB + 8, &Bs[0][brow * KP + bcb + 8]);

    for (int kt = 0; kt < KT; ++kt) {
        const int cur = kt & 1;
        const bool more = (kt + 1) < KT;
        if (more) {
            const int nxt = cur ^ 1;
            const int ko = (kt + 1) * BK;
            async_b128(gA + ko,     &As[nxt][arow * KP + acb]);
            async_b128(gA + ko + 8, &As[nxt][arow * KP + acb + 8]);
            async_b128(gB + ko,     &Bs[nxt][brow * KP + bcb]);
            if (BN == 128) async_b128(gB + ko + 8, &Bs[nxt][brow * KP + bcb + 8]);
            // wait for tile kt only; tile kt+1 stays in flight
            if (BN == 128) asm volatile("s_wait_asynccnt 4" ::: "memory");
            else           asm volatile("s_wait_asynccnt 3" ::: "memory");
        } else {
            wait_async0();
        }
        __syncthreads();   // tile kt visible to all waves

        FragB af[MT], bf_[NT];
#pragma unroll
        for (int mt = 0; mt < MT; ++mt) {
            const int b0 = (mrow0 + mt * 16 + r16) * KP + kbA;
            af[mt].q[0] = *(const uint4*)&As[cur][b0];
            af[mt].q[1] = *(const uint4*)&As[cur][b0 + 16];
        }
#pragma unroll
        for (int nt = 0; nt < NT; ++nt) {
            const int b0 = (ncol0 + nt * 16 + r16) * KP + kbB;
            bf_[nt].q[0] = *(const uint4*)&Bs[cur][b0];
            bf_[nt].q[1] = *(const uint4*)&Bs[cur][b0 + 16];
        }
#pragma unroll
        for (int mt = 0; mt < MT; ++mt)
#pragma unroll
            for (int nt = 0; nt < NT; ++nt)
                acc[mt][nt] = __builtin_amdgcn_wmma_f32_16x16x32_bf16(
                    false, af[mt].v, false, bf_[nt].v, (short)0, acc[mt][nt], false, false);
        __syncthreads();   // done reading buffer `cur` before it is re-filled
    }

    // ---- epilogue ----
#pragma unroll
    for (int mt = 0; mt < MT; ++mt)
#pragma unroll
        for (int nt = 0; nt < NT; ++nt)
#pragma unroll
            for (int r = 0; r < 8; ++r) {
                const int row  = blockRow + mrow0 + mt * 16 + r + ((lane >> 4) << 3);
                const int col  = ncol0 + nt * 16 + (lane & 15);
                const int gcol = colBase + col;
                float v = acc[mt][nt][r];
                if (HAS_BIAS) v += bias[gcol];
                if (HAS_RES)  v += res[(size_t)row * ldc + gcol];
                if (RELU)     v = fmaxf(v, 0.0f);
                if (SMODE == SM_F32) {
                    ((float*)outv)[(size_t)row * ldc + gcol] = v;
                } else if (SMODE == SM_BF16) {
                    ((ushort_t*)outv)[(size_t)row * ldc + gcol] = f2bf(v);
                } else {
                    const int h = gcol >> 6, kd = gcol & 63;
                    const int b = row >> 11, s = row & (SEQ - 1);
                    const float sv = v * storeScale;    // softmax scale folded into Q
                    if (SMODE == SM_QK)
                        ((ushort_t*)outv)[((size_t)((b * NHEADS + h) * SEQ + s)) * KDIM + kd] = f2bf(sv);
                    else  // SM_V: [bh][kd][s]
                        ((ushort_t*)outv)[((size_t)((b * NHEADS + h) * KDIM + kd)) * SEQ + s] = f2bf(sv);
                }
            }
}

// =====================================================================
// Flash attention, bf16 inputs staged by async DMA.
//   Q (pre-scaled by 1/sqrt(KD)), K: [bh][s][64] bf16; V: [bh][kd][s] bf16.
//   One block per (bh, 64-query block), 128 threads (4 waves x 16 rows).
//   Causal mask applied only on the diagonal key block.
//   Output: bf16 t0[b*S+s][h*64+kd]
// =====================================================================
__global__ __launch_bounds__(128)
void attn_kernel(const ushort_t* __restrict__ Qg, const ushort_t* __restrict__ Kg,
                 const ushort_t* __restrict__ Vg, ushort_t* __restrict__ out)
{
    constexpr int QP = 88;   // padded stride: 176B, 16B-aligned
    __shared__ __align__(16) ushort_t Qs[64 * QP];
    __shared__ __align__(16) ushort_t Ks[64 * QP];
    __shared__ __align__(16) ushort_t Vt[64 * QP];   // [kd][key]
    __shared__ __align__(16) ushort_t Ps[64 * QP];

    const int tid  = threadIdx.x;
    const int lane = tid & 31;
    const int w    = tid >> 5;
    const int bh   = blockIdx.y;
    const int qBase = blockIdx.x * 64;
    const size_t base = (size_t)bh * SEQ;

    // ---- stage Q via async DMA ----
    {
        const int r = tid >> 1, cb = (tid & 1) * 32;
        const ushort_t* g = Qg + (base + qBase + r) * KDIM + cb;
#pragma unroll
        for (int j = 0; j < 32; j += 8)
            async_b128(g + j, &Qs[r * QP + cb + j]);
    }
    wait_async0();
    __syncthreads();

    const int r16 = lane & 15;
    const int kbA = (lane >> 4) * 8;
    const int kbB = (lane >> 4) * 8;

    FragB qf[2];
#pragma unroll
    for (int c = 0; c < 2; ++c) {
        const int b0 = (w * 16 + r16) * QP + c * 32 + kbA;
        qf[c].q[0] = *(const uint4*)&Qs[b0];
        qf[c].q[1] = *(const uint4*)&Qs[b0 + 16];
    }

    v8f oacc[4];
#pragma unroll
    for (int t = 0; t < 4; ++t)
#pragma unroll
        for (int r = 0; r < 8; ++r) oacc[t][r] = 0.0f;
    float mrow[8], lrow[8];
#pragma unroll
    for (int r = 0; r < 8; ++r) { mrow[r] = -INFINITY; lrow[r] = 0.0f; }

    const int nKB = blockIdx.x + 1;   // causal
    for (int kb = 0; kb < nKB; ++kb) {
        const int kBase = kb * 64;
        __syncthreads();   // previous readers of Ks/Vt done
        {
            const int r = tid >> 1, cb = (tid & 1) * 32;
            const ushort_t* gk = Kg + (base + kBase + r) * KDIM + cb;                 // [key][kd]
            const ushort_t* gv = Vg + ((size_t)bh * KDIM + r) * SEQ + kBase + cb;     // [kd][key]
#pragma unroll
            for (int j = 0; j < 32; j += 8) {
                async_b128(gk + j, &Ks[r * QP + cb + j]);
                async_b128(gv + j, &Vt[r * QP + cb + j]);
            }
        }
        wait_async0();
        __syncthreads();

        // ---- scores S = Q K^T (Q pre-scaled) ----
        v8f sc[4];
#pragma unroll
        for (int nt = 0; nt < 4; ++nt)
#pragma unroll
            for (int r = 0; r < 8; ++r) sc[nt][r] = 0.0f;
#pragma unroll
        for (int nt = 0; nt < 4; ++nt)
#pragma unroll
            for (int c = 0; c < 2; ++c) {
                FragB kf;
                const int b0 = (nt * 16 + r16) * QP + c * 32 + kbB;
                kf.q[0] = *(const uint4*)&Ks[b0];
                kf.q[1] = *(const uint4*)&Ks[b0 + 16];
                sc[nt] = __builtin_amdgcn_wmma_f32_16x16x32_bf16(
                    false, qf[c].v, false, kf.v, (short)0, sc[nt], false, false);
            }

        const int half8 = (lane >> 4) << 3;

        // ---- causal mask: only the diagonal key block needs it ----
        if (kb == nKB - 1) {
#pragma unroll
            for (int r = 0; r < 8; ++r) {
                const int grow = qBase + w * 16 + r + half8;
#pragma unroll
                for (int nt = 0; nt < 4; ++nt) {
                    const int key = kBase + nt * 16 + (lane & 15);
                    if (key > grow) sc[nt][r] = -1e30f;
                }
            }
        }

        // ---- online softmax ----
#pragma unroll
        for (int r = 0; r < 8; ++r) {
            float mx = fmaxf(fmaxf(sc[0][r], sc[1][r]), fmaxf(sc[2][r], sc[3][r]));
            mx = fmaxf(mx, __shfl_xor(mx, 1, 32));
            mx = fmaxf(mx, __shfl_xor(mx, 2, 32));
            mx = fmaxf(mx, __shfl_xor(mx, 4, 32));
            mx = fmaxf(mx, __shfl_xor(mx, 8, 32));
            const float mnew = fmaxf(mrow[r], mx);
            const float corr = __expf(mrow[r] - mnew);
            float rsum = 0.0f;
#pragma unroll
            for (int nt = 0; nt < 4; ++nt) {
                const float p = __expf(sc[nt][r] - mnew);
                sc[nt][r] = p;
                rsum += p;
            }
            rsum += __shfl_xor(rsum, 1, 32);
            rsum += __shfl_xor(rsum, 2, 32);
            rsum += __shfl_xor(rsum, 4, 32);
            rsum += __shfl_xor(rsum, 8, 32);
            lrow[r] = lrow[r] * corr + rsum;
            mrow[r] = mnew;
#pragma unroll
            for (int t = 0; t < 4; ++t) oacc[t][r] *= corr;
        }

        // ---- P -> LDS (wave-private rows) -> A-layout frags; O += P*V ----
#pragma unroll
        for (int r = 0; r < 8; ++r)
#pragma unroll
            for (int nt = 0; nt < 4; ++nt)
                Ps[(w * 16 + r + half8) * QP + nt * 16 + (lane & 15)] = f2bf(sc[nt][r]);
        asm volatile("s_wait_dscnt 0x0" ::: "memory");

        FragB pf[2];
#pragma unroll
        for (int c = 0; c < 2; ++c) {
            const int b0 = (w * 16 + r16) * QP + c * 32 + kbA;
            pf[c].q[0] = *(const uint4*)&Ps[b0];
            pf[c].q[1] = *(const uint4*)&Ps[b0 + 16];
        }
#pragma unroll
        for (int t = 0; t < 4; ++t)
#pragma unroll
            for (int c = 0; c < 2; ++c) {
                FragB vf;
                const int b0 = (t * 16 + r16) * QP + c * 32 + kbB;
                vf.q[0] = *(const uint4*)&Vt[b0];
                vf.q[1] = *(const uint4*)&Vt[b0 + 16];
                oacc[t] = __builtin_amdgcn_wmma_f32_16x16x32_bf16(
                    false, pf[c].v, false, vf.v, (short)0, oacc[t], false, false);
            }
    }

    // ---- normalize, write bf16 concat layout ----
    const int b = bh >> 4, h = bh & 15;
    const int half8 = (lane >> 4) << 3;
#pragma unroll
    for (int t = 0; t < 4; ++t)
#pragma unroll
        for (int r = 0; r < 8; ++r) {
            const int srow = qBase + w * 16 + r + half8;
            const float val = oacc[t][r] / lrow[r];
            out[(size_t)(b * SEQ + srow) * DMODEL + h * KDIM + t * 16 + (lane & 15)] = f2bf(val);
        }
}

// =====================================================================
// LayerNorm: one block per row; writes f32 (residual stream) + bf16 (GEMM A)
// =====================================================================
__global__ __launch_bounds__(256)
void layernorm_kernel(const float* __restrict__ in, const float* __restrict__ g,
                      const float* __restrict__ b, float* __restrict__ outp,
                      ushort_t* __restrict__ outb)
{
    __shared__ float red[16];
    __shared__ float stat[2];
    const int row = blockIdx.x, tid = threadIdx.x;
    const float* x = in + (size_t)row * DMODEL;
    float s = 0.0f, ss = 0.0f;
    for (int i = tid; i < DMODEL; i += 256) { const float v = x[i]; s += v; ss += v * v; }
#pragma unroll
    for (int m = 16; m >= 1; m >>= 1) { s += __shfl_xor(s, m, 32); ss += __shfl_xor(ss, m, 32); }
    if ((tid & 31) == 0) { red[(tid >> 5) * 2] = s; red[(tid >> 5) * 2 + 1] = ss; }
    __syncthreads();
    if (tid == 0) {
        float a = 0.0f, c = 0.0f;
        for (int wv = 0; wv < 8; ++wv) { a += red[wv * 2]; c += red[wv * 2 + 1]; }
        const float mean = a * (1.0f / DMODEL);
        const float var  = c * (1.0f / DMODEL) - mean * mean;
        stat[0] = mean;
        stat[1] = rsqrtf(var + LN_EPS);
    }
    __syncthreads();
    const float mean = stat[0], rstd = stat[1];
    for (int i = tid; i < DMODEL; i += 256) {
        const float v = (x[i] - mean) * rstd * g[i] + b[i];
        outp[(size_t)row * DMODEL + i] = v;
        outb[(size_t)row * DMODEL + i] = f2bf(v);
    }
}

// =====================================================================
// Embedding gather: writes f32 (residual) + bf16 (QKV GEMM A)
// =====================================================================
__global__ __launch_bounds__(256)
void embed_kernel(const int* __restrict__ tokens, const float* __restrict__ emb,
                  float* __restrict__ x, ushort_t* __restrict__ xb)
{
    const int row = blockIdx.x;
    const size_t t = (size_t)tokens[row];
    for (int i = threadIdx.x; i < DMODEL; i += 256) {
        const float v = emb[t * DMODEL + i];
        x[(size_t)row * DMODEL + i] = v;
        xb[(size_t)row * DMODEL + i] = f2bf(v);
    }
}

// =====================================================================
// Host launcher
// =====================================================================
extern "C" void kernel_launch(void* const* d_in, const int* in_sizes, int n_in,
                              void* d_out, int out_size, void* d_ws, size_t ws_size,
                              hipStream_t stream) {
    (void)in_sizes; (void)n_in; (void)out_size; (void)ws_size;
    const int*   tokens = (const int*)  d_in[0];
    const float* emb    = (const float*)d_in[1];
    const float* Wq     = (const float*)d_in[2];
    const float* Wk     = (const float*)d_in[3];
    const float* Wv     = (const float*)d_in[4];
    const float* Wo     = (const float*)d_in[5];
    const float* bo     = (const float*)d_in[6];
    const float* g1     = (const float*)d_in[7];
    const float* b1     = (const float*)d_in[8];
    const float* Wf1    = (const float*)d_in[9];
    const float* bf1    = (const float*)d_in[10];
    const float* Wf2    = (const float*)d_in[11];
    const float* bf2    = (const float*)d_in[12];
    const float* gf     = (const float*)d_in[13];
    const float* bfb    = (const float*)d_in[14];
    const float* Wlog   = (const float*)d_in[15];
    const float* blog   = (const float*)d_in[16];
    float* out = (float*)d_out;

    const size_t MD  = (size_t)MTOT * DMODEL;       // 4M elements
    const size_t WSZ = (size_t)DMODEL * DMODEL;     // 1M elements per square weight

    float* x  = (float*)d_ws;
    float* t1 = x + MD;
    ushort_t* xb  = (ushort_t*)(t1 + MD);
    ushort_t* t0b = xb  + MD;
    ushort_t* qb  = t0b + MD;
    ushort_t* kb  = qb  + MD;
    ushort_t* vtb = kb  + MD;
    ushort_t* wts = vtb + MD;        // weight scratch: 6*WSZ per layer, or V*D for Wlog
    ushort_t* wq_t  = wts;
    ushort_t* wk_t  = wts + WSZ;
    ushort_t* wv_t  = wts + 2 * WSZ;
    ushort_t* wo_t  = wts + 3 * WSZ;
    ushort_t* wf1_t = wts + 4 * WSZ;
    ushort_t* wf2_t = wts + 5 * WSZ;

    embed_kernel<<<MTOT, 256, 0, stream>>>(tokens, emb, x, xb);

    const dim3 gT(DMODEL / 32, DMODEL / 32);         // weight transpose-convert
    const dim3 gQKV(DMODEL / 64, MTOT / 128);
    const dim3 gAttn(SEQ / 64, BATCH * NHEADS);
    const dim3 gSq(DMODEL / 128, MTOT / 128);

    for (int l = 0; l < L_LAYERS; ++l) {
        const size_t wOff = (size_t)l * WSZ;
        const size_t vOff = (size_t)l * DMODEL;

        tconv_kernel<true ><<<gT, 256, 0, stream>>>(Wq  + wOff, wq_t,  DMODEL, DMODEL);
        tconv_kernel<true ><<<gT, 256, 0, stream>>>(Wk  + wOff, wk_t,  DMODEL, DMODEL);
        tconv_kernel<true ><<<gT, 256, 0, stream>>>(Wv  + wOff, wv_t,  DMODEL, DMODEL);
        tconv_kernel<false><<<gT, 256, 0, stream>>>(Wo  + wOff, wo_t,  DMODEL, DMODEL);
        tconv_kernel<false><<<gT, 256, 0, stream>>>(Wf1 + wOff, wf1_t, DMODEL, DMODEL);
        tconv_kernel<false><<<gT, 256, 0, stream>>>(Wf2 + wOff, wf2_t, DMODEL, DMODEL);

        // Q/K/V projections (bf16 out; Q pre-scaled by 0.125; V stored transposed)
        gemm_bf16_wmma<64, SM_QK, false, false, false><<<gQKV, 256, 0, stream>>>(
            xb, wq_t, nullptr, nullptr, qb, DMODEL, DMODEL, 0.125f);
        gemm_bf16_wmma<64, SM_QK, false, false, false><<<gQKV, 256, 0, stream>>>(
            xb, wk_t, nullptr, nullptr, kb, DMODEL, DMODEL, 1.0f);
        gemm_bf16_wmma<64, SM_V, false, false, false><<<gQKV, 256, 0, stream>>>(
            xb, wv_t, nullptr, nullptr, vtb, DMODEL, DMODEL, 1.0f);

        // causal flash attention -> t0b (bf16, concat heads)
        attn_kernel<<<gAttn, 128, 0, stream>>>(qb, kb, vtb, t0b);

        // t1 = t0 @ Wo + bo + x ; x,xb = LN(t1)
        gemm_bf16_wmma<128, SM_F32, true, true, false><<<gSq, 256, 0, stream>>>(
            t0b, wo_t, bo + vOff, x, t1, DMODEL, DMODEL, 1.0f);
        layernorm_kernel<<<MTOT, 256, 0, stream>>>(t1, g1 + vOff, b1 + vOff, x, xb);

        // t0b = relu(x @ Wf1 + bf1) (bf16); t1 = t0 @ Wf2 + bf2 + x; x,xb = LN(t1)
        gemm_bf16_wmma<128, SM_BF16, true, false, true><<<gSq, 256, 0, stream>>>(
            xb, wf1_t, bf1 + vOff, nullptr, t0b, DMODEL, DMODEL, 1.0f);
        gemm_bf16_wmma<128, SM_F32, true, true, false><<<gSq, 256, 0, stream>>>(
            t0b, wf2_t, bf2 + vOff, x, t1, DMODEL, DMODEL, 1.0f);
        layernorm_kernel<<<MTOT, 256, 0, stream>>>(t1, gf + vOff, bfb + vOff, x, xb);
    }

    // logits: transpose-convert Wlog [D,V] -> [V,D] bf16, then GEMM into d_out
    const dim3 gTlog(VOCAB / 32, DMODEL / 32);
    tconv_kernel<false><<<gTlog, 256, 0, stream>>>(Wlog, wts, DMODEL, VOCAB);
    const dim3 gLog(VOCAB / 128, MTOT / 128);
    gemm_bf16_wmma<128, SM_F32, true, false, false><<<gLog, 256, 0, stream>>>(
        xb, wts, blog, nullptr, out, DMODEL, VOCAB, 1.0f);
}
